// SwinTransformerBlock_36790689857811
// MI455X (gfx1250) — matrix operand order repaired and checked by reading, not compiled, not run on hardware
//
#include <hip/hip_runtime.h>
#include <hip/hip_bf16.h>
#include <math.h>

// ---------------------------------------------------------------------------
// Swin Transformer block, MI455X (gfx1250, wave32, WMMA f32_16x16x32_f16,
// async global->LDS tile staging with double buffering).
//
// Pipeline:
//  1. wt_kernel x4      : transpose+convert weights f32 -> f16 [K,N]
//  2. ln1 kernel        : LayerNorm1 + cyclic shift(-3,-3) + window partition
//  3. gemm64<QkvEpi>    : (100352x128)@(128x384), bias, q*=scale
//  4. attn_kernel       : per (win,head): S=q@k^T (WMMA), +rel-pos bias
//                         +analytic shift mask, softmax, out = P@v (WMMA)
//  5. gemm64<ProjEpi>   : proj + window-reverse + roll(+3,+3) + residual
//  6. ln2 kernel        : LayerNorm2 -> f16
//  7. gemm64<Fc1Epi>    : (100352x128)@(128x512) + bias + exact GELU
//  8. gemm64<Fc2Epi>    : (100352x512)@(512x128) + bias + residual -> d_out
// ---------------------------------------------------------------------------

typedef __attribute__((ext_vector_type(16))) _Float16 v16h;
typedef __attribute__((ext_vector_type(8)))  float    v8f;
typedef __attribute__((ext_vector_type(4)))  _Float16 h4;
typedef __attribute__((ext_vector_type(4)))  int      v4i;

#define Hn   112
#define Wn   112
#define Cc   128
#define WSz  7
#define SHF  3
#define NHd  4
#define HDd  32
#define Nt   49
#define Bb   8
#define HIDn 512
#define Mrows 100352            // B * H * W  (= 2048 windows * 49 tokens)
#define NWIN  2048              // B * 256
#define SCALEQ 0.17677669529663687f  // 32^-0.5

// ------------- async global->LDS copy (CDNA5 GLOBAL_LOAD_ASYNC_TO_LDS) -----
#if defined(__has_builtin)
#if __has_builtin(__builtin_amdgcn_global_load_async_to_lds_b128) && \
    __has_builtin(__builtin_amdgcn_s_wait_asynccnt)
#define HAVE_ASYNC_LDS 1
#endif
#endif
#ifndef HAVE_ASYNC_LDS
#define HAVE_ASYNC_LDS 0
#endif

__device__ inline void cp_async16(const _Float16* g, _Float16* s) {
#if HAVE_ASYNC_LDS
  __builtin_amdgcn_global_load_async_to_lds_b128(
      (__attribute__((address_space(1))) v4i*)g,
      (__attribute__((address_space(3))) v4i*)s, 0, 0);
#else
  *(uint4*)s = *(const uint4*)g;
#endif
}

__device__ inline void cp_wait() {
#if HAVE_ASYNC_LDS
  __builtin_amdgcn_s_wait_asynccnt(0);
#endif
}

// ---------------- WMMA fragment loaders (per CDNA5 ISA VGPR layouts) -------

// A-matrix 16x32 f16: lanes 0-15 row M=lane hold K pairs {0..7,16..23};
// lanes 16-31 hold K {8..15,24..31}.
__device__ inline v16h frag_a(const _Float16* A, int lda, int row0, int k0) {
  int lane = threadIdx.x & 31;
  const _Float16* p = A + (size_t)(row0 + (lane & 15)) * lda + k0 + ((lane & 16) ? 8 : 0);
  v16h a;
#pragma unroll
  for (int i = 0; i < 8; ++i) {
    int kb = (i < 4) ? (2 * i) : (2 * i + 8);
    a[2 * i]     = p[kb];
    a[2 * i + 1] = p[kb + 1];
  }
  return a;
}

// B-matrix 32x16 f16: lane holds column N = lane&15; lanes 0-15 hold
// K {0..7,16..23}, lanes 16-31 hold K {8..15,24..31}.
__device__ inline v16h frag_b(const _Float16* B, int ldb, int k0, int col0) {
  int lane = threadIdx.x & 31;
  int kr = k0 + ((lane & 16) ? 8 : 0);
  const _Float16* p = B + (size_t)kr * ldb + col0 + (lane & 15);
  v16h b;
#pragma unroll
  for (int e = 0; e < 16; ++e) b[e] = p[(size_t)((e < 8) ? e : (e + 8)) * ldb];
  return b;
}

__device__ inline v8f wmma_f16(v16h a, v16h b, v8f c) {
  return __builtin_amdgcn_wmma_f32_16x16x32_f16(false, a, false, b, (short)0, c, false, false);
}

__device__ inline float wred_sum(float v) {
#pragma unroll
  for (int off = 16; off > 0; off >>= 1) v += __shfl_xor(v, off, 32);
  return v;
}

// ---------------- weight transpose/convert:  WT[k*N+n] = W[n*K+k] ----------
__global__ __launch_bounds__(256) void wt_kernel(const float* __restrict__ W,
                                                 _Float16* __restrict__ WT,
                                                 int K, int N) {
  int idx = blockIdx.x * 256 + threadIdx.x;
  if (idx < K * N) {
    int k = idx / N, n = idx % N;
    WT[idx] = (_Float16)W[(size_t)n * K + k];
  }
}

// ---------------- LN1 + shift + window partition ---------------------------
__global__ __launch_bounds__(128) void ln1_kernel(const float* __restrict__ x,
                                                  const float* __restrict__ w,
                                                  const float* __restrict__ b,
                                                  _Float16* __restrict__ xw) {
  int row  = blockIdx.x * 4 + (threadIdx.x >> 5);
  int lane = threadIdx.x & 31;
  int c = lane * 4;
  float4 v = *(const float4*)(x + (size_t)row * Cc + c);
  float s  = v.x + v.y + v.z + v.w;
  float sq = v.x * v.x + v.y * v.y + v.z * v.z + v.w * v.w;
  s = wred_sum(s); sq = wred_sum(sq);
  float mean = s * (1.0f / Cc);
  float var  = sq * (1.0f / Cc) - mean * mean;
  float inv  = rsqrtf(var + 1e-5f);
  float y0 = (v.x - mean) * inv * w[c + 0] + b[c + 0];
  float y1 = (v.y - mean) * inv * w[c + 1] + b[c + 1];
  float y2 = (v.z - mean) * inv * w[c + 2] + b[c + 2];
  float y3 = (v.w - mean) * inv * w[c + 3] + b[c + 3];
  // shifted coordinates: h' = (h - 3) mod 112
  int bb = row / (Hn * Wn);
  int rem = row % (Hn * Wn);
  int hh = rem / Wn, ww = rem % Wn;
  int hs = hh - SHF; if (hs < 0) hs += Hn;
  int ws_ = ww - SHF; if (ws_ < 0) ws_ += Wn;
  int wi = bb * 256 + (hs / WSz) * 16 + (ws_ / WSz);
  int tk = (hs % WSz) * WSz + (ws_ % WSz);
  h4 o = { (_Float16)y0, (_Float16)y1, (_Float16)y2, (_Float16)y3 };
  *(h4*)(xw + ((size_t)wi * Nt + tk) * Cc + c) = o;
}

// ---------------- LN2 (row-major, no shift) --------------------------------
__global__ __launch_bounds__(128) void ln2_kernel(const float* __restrict__ x,
                                                  const float* __restrict__ w,
                                                  const float* __restrict__ b,
                                                  _Float16* __restrict__ y) {
  int row  = blockIdx.x * 4 + (threadIdx.x >> 5);
  int lane = threadIdx.x & 31;
  int c = lane * 4;
  float4 v = *(const float4*)(x + (size_t)row * Cc + c);
  float s  = v.x + v.y + v.z + v.w;
  float sq = v.x * v.x + v.y * v.y + v.z * v.z + v.w * v.w;
  s = wred_sum(s); sq = wred_sum(sq);
  float mean = s * (1.0f / Cc);
  float var  = sq * (1.0f / Cc) - mean * mean;
  float inv  = rsqrtf(var + 1e-5f);
  h4 o = { (_Float16)((v.x - mean) * inv * w[c + 0] + b[c + 0]),
           (_Float16)((v.y - mean) * inv * w[c + 1] + b[c + 1]),
           (_Float16)((v.z - mean) * inv * w[c + 2] + b[c + 2]),
           (_Float16)((v.w - mean) * inv * w[c + 3] + b[c + 3]) };
  *(h4*)(y + (size_t)row * Cc + c) = o;
}

// ---------------- generic 64x64 WMMA GEMM, async double-buffered -----------
// C[M,N] = A[M,K] @ B[K,N]; A,B f16 row-major; 4 waves, each wave a 16-row
// strip x 64 cols (4 accumulators). M % 64 == 0, N % 64 == 0, K % 32 == 0.
// Tiles staged via GLOBAL_LOAD_ASYNC_TO_LDS_B128 (ASYNCcnt) with 2-deep
// LDS double buffering so prefetch overlaps the WMMA work.
template <class Epi>
__global__ __launch_bounds__(128) void gemm64(const _Float16* __restrict__ A,
                                              const _Float16* __restrict__ B,
                                              int K, int N, Epi epi) {
  __shared__ _Float16 As[2][64 * 32];
  __shared__ _Float16 Bs[2][32 * 64];
  int tid = threadIdx.x;
  int wave = tid >> 5, lane = tid & 31;
  int bm = blockIdx.x, bn = blockIdx.y;

  int ra = tid >> 1, offa = (tid & 1) * 16;      // A: 64 rows x 32 halves
  int rb = tid >> 2, offb = (tid & 3) * 16;      // B: 32 rows x 64 halves
  const _Float16* gA = A + ((size_t)(bm * 64 + ra)) * K + offa;
  const _Float16* gB = B + ((size_t)rb) * N + (size_t)bn * 64 + offb;

  auto issue = [&](int k0, int buf) {
    const _Float16* a = gA + k0;
    _Float16* sa = &As[buf][ra * 32 + offa];
    cp_async16(a, sa);
    cp_async16(a + 8, sa + 8);
    const _Float16* bsrc = gB + (size_t)k0 * N;
    _Float16* sb = &Bs[buf][rb * 64 + offb];
    cp_async16(bsrc, sb);
    cp_async16(bsrc + 8, sb + 8);
  };

  v8f acc[4] = {};
  issue(0, 0);
  cp_wait();
  __syncthreads();

  int nsteps = K / 32;
  for (int s = 0; s < nsteps; ++s) {
    int cur = s & 1;
    if (s + 1 < nsteps) issue((s + 1) * 32, cur ^ 1);   // prefetch next tile
    v16h a = frag_a(As[cur], 32, wave * 16, 0);
#pragma unroll
    for (int nt = 0; nt < 4; ++nt) {
      v16h bfr = frag_b(Bs[cur], 64, 0, nt * 16);
      acc[nt] = wmma_f16(a, bfr, acc[nt]);
    }
    cp_wait();
    __syncthreads();
  }

  int mrow = bm * 64 + wave * 16 + ((lane & 16) ? 8 : 0);
  int ncol = bn * 64 + (lane & 15);
#pragma unroll
  for (int nt = 0; nt < 4; ++nt)
#pragma unroll
    for (int r = 0; r < 8; ++r)
      epi(mrow + r, ncol + nt * 16, acc[nt][r]);
}

// ---------------- epilogue functors ----------------------------------------
struct QkvEpi {
  const float* bias; _Float16 *q, *k, *v;
  __device__ void operator()(int m, int n, float val) const {
    int w = m / Nt, t = m % Nt;
    int which = n >> 7, rem = n & 127, head = rem >> 5, d = rem & 31;
    float o = val + bias[n];
    size_t idx = ((size_t)(w * NHd + head) * Nt + t) * HDd + d;
    if (which == 0)      q[idx] = (_Float16)(o * SCALEQ);
    else if (which == 1) k[idx] = (_Float16)o;
    else                 v[idx] = (_Float16)o;
  }
};

struct ProjEpi {
  const float* bias; const float* shortcut; float* xnew;
  __device__ void operator()(int m, int n, float val) const {
    int w = m / Nt, t = m % Nt;
    int b = w >> 8, nw = w & 255;
    int wh = nw >> 4, ww = nw & 15;
    int th = t / WSz, tw = t % WSz;
    int hs = wh * WSz + th, ws_ = ww * WSz + tw;     // shifted coords
    int hh = hs + SHF; if (hh >= Hn) hh -= Hn;       // roll back (+3,+3)
    int w2 = ws_ + SHF; if (w2 >= Wn) w2 -= Wn;
    size_t pos = (((size_t)b * Hn + hh) * Wn + w2) * Cc + n;
    xnew[pos] = shortcut[pos] + val + bias[n];
  }
};

struct Fc1Epi {
  const float* bias; _Float16* y1;
  __device__ void operator()(int m, int n, float val) const {
    float v = val + bias[n];
    float g = 0.5f * v * (1.0f + erff(v * 0.70710678118654752f));
    y1[(size_t)m * HIDn + n] = (_Float16)g;
  }
};

struct Fc2Epi {
  const float* bias; const float* xnew; float* out;
  __device__ void operator()(int m, int n, float val) const {
    size_t i = (size_t)m * Cc + n;
    out[i] = xnew[i] + val + bias[n];
  }
};

// ---------------- fused windowed attention ---------------------------------
// One block per (window, head): S = q@k^T (4x4 WMMA tiles, 49 padded to 64),
// + relative-position bias + analytic shift mask, softmax, out = P@v.
__global__ __launch_bounds__(128) void attn_kernel(const _Float16* __restrict__ q,
                                                   const _Float16* __restrict__ k,
                                                   const _Float16* __restrict__ v,
                                                   const float* __restrict__ rpb,
                                                   _Float16* __restrict__ ao) {
  __shared__ _Float16 qs[64 * 32];
  __shared__ _Float16 ksT[32 * 64];
  __shared__ _Float16 vs[64 * 32];
  __shared__ float    Sf[64 * 64];
  __shared__ _Float16 Ph[64 * 64];
  int tid = threadIdx.x;
  int wh_id = blockIdx.x;                 // w*4 + head
  int w = wh_id >> 2, head = wh_id & 3;
  size_t base = (size_t)wh_id * Nt * HDd;
  for (int idx = tid; idx < 64 * 32; idx += 128) {
    int t = idx >> 5, d = idx & 31;
    _Float16 qv = (_Float16)0.f, kv = (_Float16)0.f, vv = (_Float16)0.f;
    if (t < Nt) {
      qv = q[base + t * HDd + d];
      kv = k[base + t * HDd + d];
      vv = v[base + t * HDd + d];
    }
    qs[idx] = qv; vs[idx] = vv; ksT[d * 64 + t] = kv;
  }
  __syncthreads();
  int wave = tid >> 5, lane = tid & 31;
  {
    v16h a = frag_a(qs, 32, wave * 16, 0);
    int mr = wave * 16 + ((lane & 16) ? 8 : 0);
    int nc = lane & 15;
#pragma unroll
    for (int nt = 0; nt < 4; ++nt) {
      v8f c = {};
      v16h bfr = frag_b(ksT, 64, 0, nt * 16);
      c = wmma_f16(a, bfr, c);
#pragma unroll
      for (int r = 0; r < 8; ++r) Sf[(mr + r) * 64 + nc + nt * 16] = c[r];
    }
  }
  __syncthreads();
  // per-row softmax with fused rel-pos bias + shift mask (rows 0..48 valid)
  if (tid < 64) {
    int i = tid;
    if (i < Nt) {
      int nw = w & 255, wwh = nw >> 4, www = nw & 15;
      int ih = i / WSz, iw = i % WSz;
      int hi_ = wwh * WSz + ih, wi_ = www * WSz + iw;
      int gh_i = (hi_ < Hn - WSz) ? 0 : (hi_ < Hn - SHF ? 1 : 2);
      int gw_i = (wi_ < Wn - WSz) ? 0 : (wi_ < Wn - SHF ? 1 : 2);
      float mx = -1e30f;
      for (int j = 0; j < Nt; ++j) {
        int jh = j / WSz, jw = j % WSz;
        int hj = wwh * WSz + jh, wj = www * WSz + jw;
        int gh_j = (hj < Hn - WSz) ? 0 : (hj < Hn - SHF ? 1 : 2);
        int gw_j = (wj < Wn - WSz) ? 0 : (wj < Wn - SHF ? 1 : 2);
        float mval = (gh_i == gh_j && gw_i == gw_j) ? 0.f : -100.f;
        int rpi = (ih - jh + WSz - 1) * (2 * WSz - 1) + (iw - jw + WSz - 1);
        float val = Sf[i * 64 + j] + rpb[rpi * NHd + head] + mval;
        Sf[i * 64 + j] = val;
        mx = fmaxf(mx, val);
      }
      float sum = 0.f;
      for (int j = 0; j < Nt; ++j) {
        float e = __expf(Sf[i * 64 + j] - mx);
        Sf[i * 64 + j] = e;
        sum += e;
      }
      float inv = 1.f / sum;
      for (int j = 0; j < Nt; ++j) Ph[i * 64 + j] = (_Float16)(Sf[i * 64 + j] * inv);
      for (int j = Nt; j < 64; ++j) Ph[i * 64 + j] = (_Float16)0.f;
    } else {
      for (int j = 0; j < 64; ++j) Ph[i * 64 + j] = (_Float16)0.f;
    }
  }
  __syncthreads();
  // out = P(64x64) @ v(64x32): two K-halves, two 16-col tiles
  v8f acc[2] = {};
#pragma unroll
  for (int ks = 0; ks < 2; ++ks) {
    v16h a = frag_a(Ph, 64, wave * 16, ks * 32);
#pragma unroll
    for (int nt = 0; nt < 2; ++nt) {
      v16h bfr = frag_b(vs, 32, ks * 32, nt * 16);
      acc[nt] = wmma_f16(a, bfr, acc[nt]);
    }
  }
  int mr = wave * 16 + ((lane & 16) ? 8 : 0);
#pragma unroll
  for (int nt = 0; nt < 2; ++nt) {
#pragma unroll
    for (int r = 0; r < 8; ++r) {
      int token = mr + r;
      if (token < Nt) {
        int d = nt * 16 + (lane & 15);
        ao[((size_t)w * Nt + token) * Cc + head * HDd + d] = (_Float16)acc[nt][r];
      }
    }
  }
}

// ---------------------------------------------------------------------------
extern "C" void kernel_launch(void* const* d_in, const int* in_sizes, int n_in,
                              void* d_out, int out_size, void* d_ws, size_t ws_size,
                              hipStream_t stream) {
  const float* x     = (const float*)d_in[0];
  const float* n1w   = (const float*)d_in[2];
  const float* n1b   = (const float*)d_in[3];
  const float* qkvw  = (const float*)d_in[4];
  const float* qkvb  = (const float*)d_in[5];
  const float* rpb   = (const float*)d_in[6];
  const float* projw = (const float*)d_in[7];
  const float* projb = (const float*)d_in[8];
  const float* n2w   = (const float*)d_in[9];
  const float* n2b   = (const float*)d_in[10];
  const float* f1w   = (const float*)d_in[11];
  const float* f1b   = (const float*)d_in[12];
  const float* f2w   = (const float*)d_in[13];
  const float* f2b   = (const float*)d_in[14];
  float* out = (float*)d_out;

  char* ws = (char*)d_ws;
  size_t off = 0;
  auto alloc = [&](size_t bytes) -> void* {
    void* p = ws + off;
    off = (off + bytes + 255) & ~(size_t)255;
    return p;
  };
  const size_t MC_h = (size_t)Mrows * Cc * sizeof(_Float16);   // 25.7 MB
  _Float16* xw_h  = (_Float16*)alloc(MC_h);
  _Float16* q_h   = (_Float16*)alloc(MC_h);
  _Float16* k_h   = (_Float16*)alloc(MC_h);
  _Float16* v_h   = (_Float16*)alloc(MC_h);
  _Float16* ao_h  = (_Float16*)alloc(MC_h);
  float*    xnew  = (float*)   alloc((size_t)Mrows * Cc * sizeof(float));
  _Float16* ln2_h = (_Float16*)alloc(MC_h);
  _Float16* y1_h  = (_Float16*)alloc((size_t)Mrows * HIDn * sizeof(_Float16));
  _Float16* wqkv_t = (_Float16*)alloc((size_t)Cc * 3 * Cc * sizeof(_Float16));
  _Float16* wproj_t = (_Float16*)alloc((size_t)Cc * Cc * sizeof(_Float16));
  _Float16* wfc1_t = (_Float16*)alloc((size_t)Cc * HIDn * sizeof(_Float16));
  _Float16* wfc2_t = (_Float16*)alloc((size_t)HIDn * Cc * sizeof(_Float16));
  (void)in_sizes; (void)n_in; (void)out_size; (void)ws_size;

  // weight transpose/convert: WT[k,n] = W[n,k]
  wt_kernel<<<(Cc * 3 * Cc + 255) / 256, 256, 0, stream>>>(qkvw, wqkv_t, Cc, 3 * Cc);
  wt_kernel<<<(Cc * Cc + 255) / 256, 256, 0, stream>>>(projw, wproj_t, Cc, Cc);
  wt_kernel<<<(Cc * HIDn + 255) / 256, 256, 0, stream>>>(f1w, wfc1_t, Cc, HIDn);
  wt_kernel<<<(HIDn * Cc + 255) / 256, 256, 0, stream>>>(f2w, wfc2_t, HIDn, Cc);

  // LN1 + shift + window partition
  ln1_kernel<<<Mrows / 4, 128, 0, stream>>>(x, n1w, n1b, xw_h);

  // QKV projection
  {
    QkvEpi e{qkvb, q_h, k_h, v_h};
    gemm64<QkvEpi><<<dim3(Mrows / 64, (3 * Cc) / 64), 128, 0, stream>>>(xw_h, wqkv_t, Cc, 3 * Cc, e);
  }

  // windowed attention (2048 windows x 4 heads)
  attn_kernel<<<NWIN * NHd, 128, 0, stream>>>(q_h, k_h, v_h, rpb, ao_h);

  // output projection + window reverse + roll + residual
  {
    ProjEpi e{projb, x, xnew};
    gemm64<ProjEpi><<<dim3(Mrows / 64, Cc / 64), 128, 0, stream>>>(ao_h, wproj_t, Cc, Cc, e);
  }

  // LN2
  ln2_kernel<<<Mrows / 4, 128, 0, stream>>>(xnew, n2w, n2b, ln2_h);

  // FC1 + GELU
  {
    Fc1Epi e{f1b, y1_h};
    gemm64<Fc1Epi><<<dim3(Mrows / 64, HIDn / 64), 128, 0, stream>>>(ln2_h, wfc1_t, Cc, HIDn, e);
  }

  // FC2 + residual -> output
  {
    Fc2Epi e{f2b, xnew, out};
    gemm64<Fc2Epi><<<dim3(Mrows / 64, Cc / 64), 128, 0, stream>>>(y1_h, wfc2_t, HIDn, Cc, e);
  }
}
